// CustomFlashAttention_42210938585821
// MI455X (gfx1250) — matrix operand order repaired and checked by reading, not compiled, not run on hardware
//
#include <hip/hip_runtime.h>
#include <hip/hip_bf16.h>

// ---------------------------------------------------------------------------
// CDNA5 (gfx1250) fused attention block:
//   q/k/v = x @ W^T   (WMMA f16, f32 accum, f32->f16 convert during LDS stage)
//   causal flash attention (WMMA QK^T and PV, online softmax)
//   out = o @ w_o^T   (WMMA, f32 output; f16 A staged via async-to-LDS)
// Workspace layout (64 MB total):
//   [0,16MB)   q  f16 [H][S][128]
//   [16,32MB)  k  f16 [H][S][128]
//   [32,48MB)  vt f16 [H][128][S]   (V transposed for contiguous B-operand)
//   [48,64MB)  o  f16 [S][2048]
// ---------------------------------------------------------------------------

typedef __attribute__((ext_vector_type(16))) _Float16 v16h;
typedef __attribute__((ext_vector_type(8)))  _Float16 v8h;
typedef __attribute__((ext_vector_type(4)))  _Float16 v4h;
typedef __attribute__((ext_vector_type(8)))  float    v8f;
typedef __attribute__((ext_vector_type(4)))  int      v4i;

#define EM_Q   0
#define EM_K   1
#define EM_V   2
#define EM_OUT 3

// ---- optional gfx1250 async global->LDS copy (ASYNCcnt-tracked) -----------
#if defined(__has_builtin)
#if __has_builtin(__builtin_amdgcn_global_load_async_to_lds_b128)
#define HAVE_ASYNC_LDS 1
#endif
#endif
#ifndef HAVE_ASYNC_LDS
#define HAVE_ASYNC_LDS 0
#endif

__device__ __forceinline__ void async_copy_b128(const void* g, void* l) {
#if HAVE_ASYNC_LDS
  // Builtin signature (per hipcc diagnostic): (int4 AS1*, int4 AS3*, imm, imm)
  typedef __attribute__((address_space(1))) v4i* gp_t;
  typedef __attribute__((address_space(3))) v4i* lp_t;
  __builtin_amdgcn_global_load_async_to_lds_b128((gp_t)g, (lp_t)l, 0, 0);
#else
  *(v8h*)l = *(const v8h*)g;
#endif
}

__device__ __forceinline__ void wait_async0() {
#if HAVE_ASYNC_LDS
#if __has_builtin(__builtin_amdgcn_s_wait_asynccnt)
  __builtin_amdgcn_s_wait_asynccnt(0);
#else
  asm volatile("s_wait_asynccnt 0" ::: "memory");
#endif
#endif
}

// ---------------------------------------------------------------------------
__device__ __forceinline__ v8f wmma_f16(v16h a, v16h b, v8f c) {
  // v_wmma_f32_16x16x32_f16: D = A(16x32 f16) * B(32x16 f16) + C(16x16 f32)
  return __builtin_amdgcn_wmma_f32_16x16x32_f16(false, a, false, b, (short)0, c,
                                                false, false);
}

__device__ __forceinline__ v16h pack16(v8h lo, v8h hi) {
  v16h a;
#pragma unroll
  for (int i = 0; i < 8; ++i) { a[i] = lo[i]; a[i + 8] = hi[i]; }
  return a;
}

// A-matrix 16x32 f16 fragment: lanes 0-15 hold rows M=0..15 with
// K = 0..15; lanes 16-31 hold the same rows with K = 16..31.
__device__ __forceinline__ v16h load_afrag(const _Float16* base, int ld) {
  int lane = threadIdx.x & 31;
  int row  = lane & 15;
  int koff = (lane >> 4) ? 16 : 0;
  const _Float16* p = base + (size_t)row * ld + koff;
  return pack16(*(const v8h*)p, *(const v8h*)(p + 8));
}

// B-matrix 32x16 f16 fragment: lane n (0..15) holds column N=n; lanes 0-15
// carry K=0..15, lanes 16-31 carry K=16..31. `base` points at column 0 of a
// column-major tile (rows of W / rows of K / rows of V^T, contiguous along K).
__device__ __forceinline__ v16h load_bfrag(const _Float16* base, int ld) {
  int lane = threadIdx.x & 31;
  int col  = lane & 15;
  int koff = (lane >> 4) ? 16 : 0;
  const _Float16* p = base + (size_t)col * ld + koff;
  return pack16(*(const v8h*)p, *(const v8h*)(p + 8));
}

// ---------------------------------------------------------------------------
// Tile staging: A (f32->f16 convert, or raw f16 async copy) and B (f32->f16).
// ---------------------------------------------------------------------------
template <int AF16>
__device__ __forceinline__ void stage_tiles(const void* __restrict__ Av,
                                            const float* __restrict__ Bw,
                                            int m0, int n0, int K, int kb,
                                            _Float16 (*As)[32],
                                            _Float16 (*Bs)[32], int tid) {
  if (AF16) {
    const _Float16* A = (const _Float16*)Av;
#pragma unroll
    for (int it = 0; it < 2; ++it) {          // 512 x 16B chunks / 256 threads
      int idx = tid + it * 256;
      int row = idx >> 2, ch = idx & 3;       // 4 chunks of 8 halves per row
      async_copy_b128(A + (size_t)(m0 + row) * K + kb + ch * 8,
                      &As[row][ch * 8]);
    }
  } else {
    const float* A = (const float*)Av;
#pragma unroll
    for (int it = 0; it < 4; ++it) {          // 1024 float4 chunks
      int idx = tid + it * 256;
      int row = idx >> 3, ch = idx & 7;       // 8 chunks of 4 floats per row
      float4 f = *(const float4*)(A + (size_t)(m0 + row) * K + kb + ch * 4);
      v4h hv; hv[0] = (_Float16)f.x; hv[1] = (_Float16)f.y;
      hv[2] = (_Float16)f.z; hv[3] = (_Float16)f.w;
      *(v4h*)&As[row][ch * 4] = hv;
    }
  }
#pragma unroll
  for (int it = 0; it < 4; ++it) {
    int idx = tid + it * 256;
    int row = idx >> 3, ch = idx & 7;
    float4 f = *(const float4*)(Bw + (size_t)(n0 + row) * K + kb + ch * 4);
    v4h hv; hv[0] = (_Float16)f.x; hv[1] = (_Float16)f.y;
    hv[2] = (_Float16)f.z; hv[3] = (_Float16)f.w;
    *(v4h*)&Bs[row][ch * 4] = hv;
  }
}

// ---------------------------------------------------------------------------
// C = A[M,K] @ Bw[N,K]^T  (out[m,n] = sum_k A[m,k]*Bw[n,k])
// 128x128 tile / 256-thread block (8 waves), K-step 32, double-buffered LDS.
// ---------------------------------------------------------------------------
template <int AF16, int EMODE>
__global__ __launch_bounds__(256)
void gemm_xwT_kernel(const void* __restrict__ Av, const float* __restrict__ Bw,
                     void* __restrict__ Dst, int M, int N, int K) {
  const int tid  = threadIdx.x;
  const int lane = tid & 31;
  const int wv   = tid >> 5;      // 0..7
  const int wm   = wv & 3;        // 4 waves along M
  const int wn   = wv >> 2;       // 2 waves along N
  const int m0   = blockIdx.y * 128;
  const int n0   = blockIdx.x * 128;

  __shared__ _Float16 As[2][128][32];
  __shared__ _Float16 Bs[2][128][32];

  v8f acc[2][4];
#pragma unroll
  for (int mi = 0; mi < 2; ++mi)
#pragma unroll
    for (int ni = 0; ni < 4; ++ni) acc[mi][ni] = (v8f){};

  const int nk = K >> 5;
  stage_tiles<AF16>(Av, Bw, m0, n0, K, 0, As[0], Bs[0], tid);
  if (AF16) wait_async0();
  __syncthreads();

  for (int t = 0; t < nk; ++t) {
    const int cur = t & 1;

    // fragments for current tile (ds_load_b128)
    v16h af[2], bf[4];
#pragma unroll
    for (int mi = 0; mi < 2; ++mi)
      af[mi] = load_afrag(&As[cur][wm * 32 + mi * 16][0], 32);
#pragma unroll
    for (int ni = 0; ni < 4; ++ni)
      bf[ni] = load_bfrag(&Bs[cur][wn * 64 + ni * 16][0], 32);

    // prefetch next tile into the other buffer while WMMAs run
    if (t + 1 < nk)
      stage_tiles<AF16>(Av, Bw, m0, n0, K, (t + 1) * 32, As[cur ^ 1],
                        Bs[cur ^ 1], tid);

#pragma unroll
    for (int mi = 0; mi < 2; ++mi)
#pragma unroll
      for (int ni = 0; ni < 4; ++ni)
        acc[mi][ni] = wmma_f16(af[mi], bf[ni], acc[mi][ni]);

    if (AF16) wait_async0();
    __syncthreads();
  }

  // ---- epilogue: C layout M = r + 8*(lane>=16), N = lane&15 ----
  const int hi8 = (lane >> 4) * 8;
  const int nl  = lane & 15;
#pragma unroll
  for (int mi = 0; mi < 2; ++mi) {
#pragma unroll
    for (int ni = 0; ni < 4; ++ni) {
#pragma unroll
      for (int r = 0; r < 8; ++r) {
        int   gm  = m0 + wm * 32 + mi * 16 + r + hi8;
        int   gn  = n0 + wn * 64 + ni * 16 + nl;
        float val = acc[mi][ni][r];
        if (EMODE == EM_OUT) {
          ((float*)Dst)[(size_t)gm * N + gn] = val;
        } else {
          _Float16* dst = (_Float16*)Dst;
          int hd = gn >> 7, dh = gn & 127;
          if (EMODE == EM_V)        // V stored transposed: [H][Dh][S]
            dst[((size_t)hd * 128 + dh) * M + gm] = (_Float16)val;
          else                      // Q/K stored [H][S][Dh]
            dst[((size_t)hd * M + gm) * 128 + dh] = (_Float16)val;
        }
      }
    }
  }
}

// ---------------------------------------------------------------------------
// Causal flash attention. Block = 128 threads (4 waves), each wave owns a
// 16-row q tile; grid = (S/64, H). Online softmax in f32, matmuls in WMMA.
// ---------------------------------------------------------------------------
__global__ __launch_bounds__(128)
void flash_attn_kernel(const void* __restrict__ qv, const void* __restrict__ kv,
                       const void* __restrict__ vtv, void* __restrict__ ov,
                       const int* __restrict__ causal_p, int S, int H) {
  const int h    = blockIdx.y;
  const int tid  = threadIdx.x;
  const int lane = tid & 31;
  const int wv   = tid >> 5;                 // 0..3
  const int q0   = blockIdx.x * 64 + wv * 16;
  const int causal = *causal_p;
  const int hi8  = (lane >> 4) * 8;
  const int nl   = lane & 15;

  const _Float16* qh = (const _Float16*)qv  + (size_t)h * S * 128;
  const _Float16* kh = (const _Float16*)kv  + (size_t)h * S * 128;
  const _Float16* vh = (const _Float16*)vtv + (size_t)h * 128 * S;
  _Float16*       o  = (_Float16*)ov;

  __shared__ _Float16 Pb[4][16 * 32];        // per-wave P staging (C -> A layout)
  _Float16* Pw = &Pb[wv][0];

  // Q fragments: 16 rows x 128 (dh) = 4 x (16x32) A-operands, loaded once.
  v16h qf[4];
  {
    int row = lane & 15;
    int koff = (lane >> 4) ? 16 : 0;
#pragma unroll
    for (int kk = 0; kk < 4; ++kk) {
      const _Float16* p = qh + (size_t)(q0 + row) * 128 + kk * 32 + koff;
      qf[kk] = pack16(*(const v8h*)p, *(const v8h*)(p + 8));
    }
  }

  v8f acc[8];
#pragma unroll
  for (int t = 0; t < 8; ++t) acc[t] = (v8f){};
  float mrow[8], lrow[8];
#pragma unroll
  for (int r = 0; r < 8; ++r) { mrow[r] = -1e30f; lrow[r] = 0.0f; }

  const float sm_scale = 0.08838834764831845f;  // 1/sqrt(128)
  const float LOG2E    = 1.4426950408889634f;
  const int   kend     = causal ? (q0 + 15) : (S - 1);

  for (int k0 = 0; k0 <= kend; k0 += 32) {
    // ---- scores: two 16-wide column tiles, K-dim = dh (4 x 32) ----
    v8f sc[2];
#pragma unroll
    for (int half = 0; half < 2; ++half) {
      v8f s = (v8f){};
      int kcol0 = k0 + half * 16;
      int koff  = (lane >> 4) ? 16 : 0;
#pragma unroll
      for (int kk = 0; kk < 4; ++kk) {
        const _Float16* p = kh + (size_t)(kcol0 + nl) * 128 + kk * 32 + koff;
        v16h bf = pack16(*(const v8h*)p, *(const v8h*)(p + 8));
        s = wmma_f16(qf[kk], bf, s);
      }
      sc[half] = s;
    }

    // ---- scale + causal mask + row max ----
    float tmax[8];
#pragma unroll
    for (int r = 0; r < 8; ++r) {
      float s0 = sc[0][r] * sm_scale;
      float s1 = sc[1][r] * sm_scale;
      int qrow = q0 + r + hi8;
      if (causal) {
        if (qrow < k0 + nl)      s0 = -1e30f;
        if (qrow < k0 + 16 + nl) s1 = -1e30f;
      }
      sc[0][r] = s0; sc[1][r] = s1;
      tmax[r]  = fmaxf(s0, s1);
    }
#pragma unroll
    for (int r = 0; r < 8; ++r)
#pragma unroll
      for (int off = 1; off < 16; off <<= 1)
        tmax[r] = fmaxf(tmax[r], __shfl_xor(tmax[r], off, 32));

    // ---- online softmax update ----
    float alpha[8], psum[8];
#pragma unroll
    for (int r = 0; r < 8; ++r) {
      float nm = fmaxf(mrow[r], tmax[r]);
      alpha[r] = __builtin_amdgcn_exp2f((mrow[r] - nm) * LOG2E);
      mrow[r]  = nm;
      float p0 = __builtin_amdgcn_exp2f((sc[0][r] - nm) * LOG2E);
      float p1 = __builtin_amdgcn_exp2f((sc[1][r] - nm) * LOG2E);
      Pw[(r + hi8) * 32 + nl]      = (_Float16)p0;   // stage P for A-operand
      Pw[(r + hi8) * 32 + 16 + nl] = (_Float16)p1;
      psum[r] = p0 + p1;
    }
#pragma unroll
    for (int r = 0; r < 8; ++r)
#pragma unroll
      for (int off = 1; off < 16; off <<= 1)
        psum[r] += __shfl_xor(psum[r], off, 32);
#pragma unroll
    for (int r = 0; r < 8; ++r) lrow[r] = lrow[r] * alpha[r] + psum[r];
#pragma unroll
    for (int t = 0; t < 8; ++t)
#pragma unroll
      for (int r = 0; r < 8; ++r) acc[t][r] *= alpha[r];

    // per-wave LDS writes above feed the read below (cross-lane): wait DScnt.
    asm volatile("s_wait_dscnt 0" ::: "memory");

    v16h pf;
    {
      int row = lane & 15;
      int koff = (lane >> 4) ? 16 : 0;
      const _Float16* p = Pw + row * 32 + koff;
      pf = pack16(*(const v8h*)p, *(const v8h*)(p + 8));
    }

    // ---- O += P @ V : 8 dh-tiles, V^T rows are contiguous along s ----
    {
      int koff = (lane >> 4) ? 16 : 0;
#pragma unroll
      for (int t = 0; t < 8; ++t) {
        const _Float16* p = vh + (size_t)(t * 16 + nl) * S + k0 + koff;
        v16h bf = pack16(*(const v8h*)p, *(const v8h*)(p + 8));
        acc[t] = wmma_f16(pf, bf, acc[t]);
      }
    }
  }

  // ---- finalize: O /= l, store f16 to o[S][H*128] ----
  float inv[8];
#pragma unroll
  for (int r = 0; r < 8; ++r) inv[r] = 1.0f / lrow[r];
#pragma unroll
  for (int t = 0; t < 8; ++t)
#pragma unroll
    for (int r = 0; r < 8; ++r) {
      int qrow = q0 + r + hi8;
      o[(size_t)qrow * (H * 128) + h * 128 + t * 16 + nl] =
          (_Float16)(acc[t][r] * inv[r]);
    }
}

// ---------------------------------------------------------------------------
extern "C" void kernel_launch(void* const* d_in, const int* in_sizes, int n_in,
                              void* d_out, int out_size, void* d_ws, size_t ws_size,
                              hipStream_t stream) {
  const float* x   = (const float*)d_in[0];
  const float* w_q = (const float*)d_in[1];
  const float* w_k = (const float*)d_in[2];
  const float* w_v = (const float*)d_in[3];
  const float* w_o = (const float*)d_in[4];
  const int* causal = (const int*)d_in[5];

  const int S = 4096, D = 2048, H = 16;
  const size_t seg = (size_t)S * D * 2;   // 16 MB per f16 [S][D]-sized buffer

  char* ws = (char*)d_ws;
  void* q  = (void*)(ws + 0 * seg);
  void* k  = (void*)(ws + 1 * seg);
  void* vt = (void*)(ws + 2 * seg);
  void* o  = (void*)(ws + 3 * seg);       // total 64 MB of workspace

  dim3 gblk(256), ggrid(D / 128, S / 128);
  gemm_xwT_kernel<0, EM_Q><<<ggrid, gblk, 0, stream>>>(x, w_q, q,  S, D, D);
  gemm_xwT_kernel<0, EM_K><<<ggrid, gblk, 0, stream>>>(x, w_k, k,  S, D, D);
  gemm_xwT_kernel<0, EM_V><<<ggrid, gblk, 0, stream>>>(x, w_v, vt, S, D, D);

  flash_attn_kernel<<<dim3(S / 64, H), 128, 0, stream>>>(q, k, vt, o, causal, S, H);

  gemm_xwT_kernel<1, EM_OUT><<<ggrid, gblk, 0, stream>>>(o, w_o, d_out, S, D, D);
}
